// AxialLongShortTransformer_68358699483522
// MI455X (gfx1250) — matrix operand order repaired and checked by reading, not compiled, not run on hardware
//
#include <hip/hip_runtime.h>
#include <hip/hip_bf16.h>

// ---------------------------------------------------------------------------
// AxialLongShortTransformer forward for gfx1250 (MI455X), wave32 + WMMA.
// All dense math runs through v_wmma_f32_16x16x32_f16 (f16 A/B, f32 acc).
// All LDS fragment traffic is 128-bit: weights are pre-transposed to N-major
// so B fragments are contiguous per lane; V is produced (s,h,d,t) so the AV
// operand is contiguous too. LayerNorm/softmax stay fp32 VALU (wave32
// shuffles for the d=32 head reductions).
// ---------------------------------------------------------------------------

typedef __attribute__((ext_vector_type(16))) _Float16 v16h;
typedef __attribute__((ext_vector_type(8)))  _Float16 v8h;
typedef __attribute__((ext_vector_type(8)))  float    v8f;

#define WI    192          // H == W == 192
#define HW    36864        // 192*192 pixels (also S*T tokens)
#define DIMC  128
#define INNER 256          // heads(8) * dim_head(32)
#define NHEAD 8
#define DHEAD 32
#define WS    64           // window size
#define NWIN  3

__device__ __forceinline__ v8f wmma32(v16h a, v16h b, v8f c) {
  return __builtin_amdgcn_wmma_f32_16x16x32_f16(false, a, false, b, (short)0, c,
                                                false, false);
}

__device__ __forceinline__ v8h v8h_zero() {
  v8h z;
#pragma unroll
  for (int e = 0; e < 8; ++e) z[e] = (_Float16)0.f;
  return z;
}

// A fragment: 16x32 f16 row-major (ISA 7.12.2): lanes 0-15 rows M, K 0-7 & 16-23;
// lanes 16-31 rows M, K 8-15 & 24-31. Two 16B vector loads per lane.
__device__ __forceinline__ v16h load_a_frag(const _Float16* A, int lda) {
  int lane = threadIdx.x & 31;
  int row  = lane & 15;
  int kb   = (lane >> 4) << 3;   // 0 or 8
  const _Float16* p = A + row * lda + kb;
  v8h lo = *(const v8h*)(p);
  v8h hi = *(const v8h*)(p + 16);
  return __builtin_shufflevector(lo, hi, 0, 1, 2, 3, 4, 5, 6, 7,
                                 8, 9, 10, 11, 12, 13, 14, 15);
}

// B fragment from N-major storage Bt[n][k] (ld = ldt): lanes 0-15 col N with
// K 0-7 & 16-23; lanes 16-31 col N with K 8-15 & 24-31. Two 16B loads per lane.
__device__ __forceinline__ v16h load_bt_frag(const _Float16* Bt, int ldt) {
  int lane = threadIdx.x & 31;
  int col  = lane & 15;
  int kb   = (lane >> 4) << 3;   // 0 or 8
  const _Float16* p = Bt + col * ldt + kb;
  v8h lo = *(const v8h*)(p);
  v8h hi = *(const v8h*)(p + 16);
  return __builtin_shufflevector(lo, hi, 0, 1, 2, 3, 4, 5, 6, 7,
                                 8, 9, 10, 11, 12, 13, 14, 15);
}

// ------------------------------ weight prep --------------------------------

// dense weight (K,N) f32 -> Wt (N,K) f16  (N-major for contiguous B frags)
__global__ __launch_bounds__(256) void k_cvt_w_T(const float* __restrict__ w,
                                                 _Float16* __restrict__ wt,
                                                 int K, int N) {
  int i = blockIdx.x * 256 + threadIdx.x;
  if (i >= K * N) return;
  int n = i / K, k = i - n * K;
  wt[i] = (_Float16)w[(size_t)k * N + n];
}

// conv weight (OC,IC,3,3) f32 -> Bmat[oc][tap*IC+ic] f16 (OC-major)
__global__ __launch_bounds__(256) void k_prep_conv_w(const float* __restrict__ w,
                                                     _Float16* __restrict__ bmat,
                                                     int IC, int OC) {
  int K = IC * 9;
  int i = blockIdx.x * 256 + threadIdx.x;
  if (i >= K * OC) return;
  int oc  = i / K;
  int k   = i - oc * K;
  int tap = k / IC;
  int ic  = k - tap * IC;
  bmat[i] = (_Float16)w[((size_t)(oc * IC + ic)) * 9 + tap];
}

// ------------------------------ small kernels ------------------------------

// axial gather + pre-norm: xcur (C,H,W) f32 -> xn16 (tok, 128) f16
__global__ __launch_bounds__(128) void k_gather_ln(const float* __restrict__ x,
                                                   const float* __restrict__ g,
                                                   const float* __restrict__ b,
                                                   _Float16* __restrict__ xn, int axis) {
  int tok = blockIdx.x;                 // s*192 + t
  int s = tok / WI, t = tok - s * WI;
  int hh = axis ? s : t;
  int ww = axis ? t : s;
  int c = threadIdx.x;
  float v = x[(size_t)c * HW + hh * WI + ww];
  __shared__ float red[128];
  red[c] = v; __syncthreads();
  for (int o = 64; o > 0; o >>= 1) { if (c < o) red[c] += red[c + o]; __syncthreads(); }
  float mu = red[0] * (1.f / 128.f);
  __syncthreads();
  float dv = v - mu;
  red[c] = dv * dv; __syncthreads();
  for (int o = 64; o > 0; o >>= 1) { if (c < o) red[c] += red[c + o]; __syncthreads(); }
  float var = red[0] * (1.f / 128.f);
  xn[(size_t)tok * DIMC + c] = (_Float16)(dv * rsqrtf(var + 1e-5f) * g[c] + b[c]);
}

// ------------------------------ dense GEMM ---------------------------------
// C(MxN,f32) = A(MxK,f16,row-major) * Bt(NxK,f16,N-major); M%64==0,N%64==0,K%32==0
__global__ __launch_bounds__(256) void k_gemm(const _Float16* __restrict__ A,
                                              const _Float16* __restrict__ Bt,
                                              float* __restrict__ C,
                                              int M, int N, int K) {
  __shared__ __align__(16) _Float16 As[64][32];
  __shared__ __align__(16) _Float16 BsT[64][32];
  int tid = threadIdx.x;
  int m0 = blockIdx.y * 64, n0 = blockIdx.x * 64;
  int waveId = tid >> 5, lane = tid & 31;
  int wm = waveId & 3, wn2 = waveId >> 2;
  v8f acc0 = {0.f,0.f,0.f,0.f,0.f,0.f,0.f,0.f};
  v8f acc1 = {0.f,0.f,0.f,0.f,0.f,0.f,0.f,0.f};
  int row8 = tid >> 2, col8 = (tid & 3) * 8;
  const _Float16* ap = A + (size_t)(m0 + row8) * K + col8;
  const _Float16* bp = Bt + (size_t)(n0 + row8) * K + col8;
  for (int k0 = 0; k0 < K; k0 += 32) {
    *(v8h*)&As[row8][col8]  = *(const v8h*)(ap + k0);
    *(v8h*)&BsT[row8][col8] = *(const v8h*)(bp + k0);
    if (k0 + 32 < K) {
      __builtin_prefetch(ap + k0 + 32, 0, 1);
      __builtin_prefetch(bp + k0 + 32, 0, 1);
    }
    __syncthreads();
    v16h a  = load_a_frag(&As[wm * 16][0], 32);
    v16h b0 = load_bt_frag(&BsT[wn2 * 32][0], 32);
    v16h b1 = load_bt_frag(&BsT[wn2 * 32 + 16][0], 32);
    acc0 = wmma32(a, b0, acc0);
    acc1 = wmma32(a, b1, acc1);
    __syncthreads();
  }
  int col = lane & 15, rb = (lane >> 4) * 8;
#pragma unroll
  for (int r = 0; r < 8; ++r) {
    int m = m0 + wm * 16 + rb + r;
    C[(size_t)m * N + n0 + wn2 * 32 + col]      = acc0[r];
    C[(size_t)m * N + n0 + wn2 * 32 + 16 + col] = acc1[r];
  }
}

// ------------------------------ implicit-GEMM conv3x3 ----------------------
// img (IC,192,192) f16; Bw (OC, 9*IC) f16 OC-major; C (36864, OC) f32. IC%32==0.
__global__ __launch_bounds__(256) void k_conv_gemm(const _Float16* __restrict__ img,
                                                   const _Float16* __restrict__ Bw,
                                                   float* __restrict__ C,
                                                   int IC, int OC) {
  __shared__ __align__(16) _Float16 As[64][32];
  __shared__ __align__(16) _Float16 BsT[64][32];
  int tid = threadIdx.x;
  int m0 = blockIdx.y * 64, n0 = blockIdx.x * 64;
  int waveId = tid >> 5, lane = tid & 31;
  int wm = waveId & 3, wn2 = waveId >> 2;
  v8f acc0 = {0.f,0.f,0.f,0.f,0.f,0.f,0.f,0.f};
  v8f acc1 = {0.f,0.f,0.f,0.f,0.f,0.f,0.f,0.f};
  int row8 = tid >> 2, col8 = (tid & 3) * 8;
  int m = m0 + row8;
  int y = m / WI, x = m - y * WI;
  int K = IC * 9;
  const _Float16* bp = Bw + (size_t)(n0 + row8) * K + col8;
  for (int k0 = 0; k0 < K; k0 += 32) {
    int tap = k0 / IC;                    // constant across a 32-chunk (IC%32==0)
    int dy = tap / 3 - 1, dx = tap - (tap / 3) * 3 - 1;
    int yy = y + dy, xx = x + dx;
    bool ok = (yy >= 0) & (yy < WI) & (xx >= 0) & (xx < WI);
    int icb = k0 - tap * IC + col8;
    v8h av = v8h_zero();
    if (ok) {
#pragma unroll
      for (int e = 0; e < 8; ++e)
        av[e] = img[(size_t)(icb + e) * HW + yy * WI + xx];
    }
    *(v8h*)&As[row8][col8]  = av;
    *(v8h*)&BsT[row8][col8] = *(const v8h*)(bp + k0);
    if (k0 + 32 < K) __builtin_prefetch(bp + k0 + 32, 0, 1);
    __syncthreads();
    v16h a  = load_a_frag(&As[wm * 16][0], 32);
    v16h b0 = load_bt_frag(&BsT[wn2 * 32][0], 32);
    v16h b1 = load_bt_frag(&BsT[wn2 * 32 + 16][0], 32);
    acc0 = wmma32(a, b0, acc0);
    acc1 = wmma32(a, b1, acc1);
    __syncthreads();
  }
  int col = lane & 15, rb = (lane >> 4) * 8;
#pragma unroll
  for (int r = 0; r < 8; ++r) {
    int mm = m0 + wm * 16 + rb + r;
    C[(size_t)mm * OC + n0 + wn2 * 32 + col]      = acc0[r];
    C[(size_t)mm * OC + n0 + wn2 * 32 + 16 + col] = acc1[r];
  }
}

// ------------------------------ attention helpers --------------------------

// q (tok,256) f32 -> q16 (s,h,t,d) f16, scaled by d^-0.5
__global__ __launch_bounds__(256) void k_scale_q(const float* __restrict__ q,
                                                 _Float16* __restrict__ q16) {
  int i = blockIdx.x * 256 + threadIdx.x;
  if (i >= HW * INNER) return;
  int tok = i >> 8, hd = i & 255;
  int s = tok / WI, t = tok - s * WI;
  int h = hd >> 5, dd = hd & 31;
  q16[(((size_t)(s * NHEAD + h) * WI) + t) * DHEAD + dd] =
      (_Float16)(q[i] * 0.17677669529663687f);
}

// kv (tok,512) f32 -> lk16 = LN_d(k) in (s,h,t,d); v16 in (s,h,d,t) f16.
__global__ __launch_bounds__(256) void k_ln_k(const float* __restrict__ kv,
                                              const float* __restrict__ g,
                                              const float* __restrict__ b,
                                              _Float16* __restrict__ lk,
                                              _Float16* __restrict__ vv) {
  int tok = blockIdx.x;
  int h = threadIdx.x >> 5, lane = threadIdx.x & 31;
  int s = tok / WI, t = tok - s * WI;
  float kvv = kv[(size_t)tok * 512 + h * DHEAD + lane];
  float vvv = kv[(size_t)tok * 512 + 256 + h * DHEAD + lane];
  float sum = kvv;
  for (int m = 16; m; m >>= 1) sum += __shfl_xor(sum, m, 32);
  float mu = sum * (1.f / 32.f);
  float d = kvv - mu, sq = d * d;
  for (int m = 16; m; m >>= 1) sq += __shfl_xor(sq, m, 32);
  float yv = d * rsqrtf(sq * (1.f / 32.f) + 1e-5f) * g[lane] + b[lane];
  int sh = s * NHEAD + h;
  lk[((size_t)sh * WI + t) * DHEAD + lane] = (_Float16)yv;        // (s,h,t,d)
  vv[((size_t)sh * DHEAD + lane) * WI + t] = (_Float16)vvv;       // (s,h,d,t)
}

// landmark: p = softmax_t(k . Wproj); gk = LN(sum p k), gv = sum p v
__global__ __launch_bounds__(192) void k_landmark(const float* __restrict__ kv,
                                                  const float* __restrict__ wproj,
                                                  const float* __restrict__ g,
                                                  const float* __restrict__ b,
                                                  float* __restrict__ gk,
                                                  float* __restrict__ gv) {
  int sh = blockIdx.x;           // s*8 + h
  int s = sh >> 3, h = sh & 7;
  int t = threadIdx.x;           // 0..191
  __shared__ float lg[192];
  __shared__ float pbuf[192];
  const float* kb = kv + (size_t)(s * WI + t) * 512 + h * DHEAD;
  float acc = 0.f;
  for (int dd = 0; dd < DHEAD; ++dd) acc += kb[dd] * wproj[dd];
  lg[t] = acc; __syncthreads();
  float mx = -1e30f;
  for (int j = 0; j < WI; ++j) mx = fmaxf(mx, lg[j]);
  float e = __expf(lg[t] - mx);
  pbuf[t] = e; __syncthreads();
  float ssum = 0.f;
  for (int j = 0; j < WI; ++j) ssum += pbuf[j];
  __syncthreads();
  pbuf[t] = e / ssum; __syncthreads();
  if (t < DHEAD) {   // threads 0..31 == wave 0
    float ak = 0.f, av = 0.f;
    for (int j = 0; j < WI; ++j) {
      const float* kj = kv + (size_t)(s * WI + j) * 512 + h * DHEAD;
      ak += pbuf[j] * kj[t];
      av += pbuf[j] * kj[256 + t];
    }
    float sum = ak;
    for (int m = 16; m; m >>= 1) sum += __shfl_xor(sum, m, 32);
    float mu = sum * (1.f / 32.f);
    float d = ak - mu, sq = d * d;
    for (int m = 16; m; m >>= 1) sq += __shfl_xor(sq, m, 32);
    gk[sh * DHEAD + t] = d * rsqrtf(sq * (1.f / 32.f) + 1e-5f) * g[t] + b[t];
    gv[sh * DHEAD + t] = av;
  }
}

// windowed attention with +-1 look-around + 1 landmark; one block per (win, s*h)
__global__ __launch_bounds__(256) void k_attn(const _Float16* __restrict__ q16,
                                              const _Float16* __restrict__ lk16,
                                              const _Float16* __restrict__ v16g,
                                              const float* __restrict__ gk,
                                              const float* __restrict__ gv,
                                              _Float16* __restrict__ attn16) {
  __shared__ __align__(16) char pool[4096 + 12288 + 24576];
  __shared__ float simg[64], pg[64], gks[32], gvs[32];
  _Float16* Qs  = (_Float16*)pool;                 // 64 x 32  (t, d)
  _Float16* Ks  = (_Float16*)(pool + 4096);        // 192 x 32 (key, d) = N-major
  _Float16* Sch = (_Float16*)(pool + 16384);       // 64 x 192 scores -> probs
  _Float16* VsT = (_Float16*)(pool + 4096);        // 32 x 192 (d, key), reuses Ks

  int win = blockIdx.x;          // 0..2
  int sh  = blockIdx.y;          // s*8 + h
  int tid = threadIdx.x;
  int waveId = tid >> 5, lane = tid & 31;
  int tbase = win * WS - WS;

  // stage Q (one 16B chunk per thread) and K (8-wide chunks, pad -> 0)
  {
    const v8h* qb = (const v8h*)(q16 + (size_t)sh * WI * DHEAD +
                                 (size_t)win * WS * DHEAD);
    ((v8h*)Qs)[tid] = qb[tid];
  }
  for (int ch = tid; ch < WI * 4; ch += 256) {
    int j = ch >> 2, dg = (ch & 3) * 8;
    int tg = tbase + j;
    v8h val = v8h_zero();
    if (tg >= 0 && tg < WI)
      val = *(const v8h*)(lk16 + ((size_t)sh * WI + tg) * DHEAD + dg);
    *(v8h*)(Ks + j * DHEAD + dg) = val;
  }
  if (tid < 32) { gks[tid] = gk[sh * DHEAD + tid]; gvs[tid] = gv[sh * DHEAD + tid]; }
  __syncthreads();

  // sim_l = Q (64x32) @ K^T (32x192): 4x12 tiles distributed over 8 waves
  for (int tIdx = waveId; tIdx < 48; tIdx += 8) {
    int tm = tIdx / 12, tn = tIdx - tm * 12;
    v16h a  = load_a_frag(Qs + tm * 16 * DHEAD, DHEAD);
    v16h bt = load_bt_frag(Ks + tn * 16 * DHEAD, DHEAD);
    v8f acc = {0.f,0.f,0.f,0.f,0.f,0.f,0.f,0.f};
    acc = wmma32(a, bt, acc);
    int col = lane & 15, rb = (lane >> 4) * 8;
#pragma unroll
    for (int r = 0; r < 8; ++r)
      Sch[(tm * 16 + rb + r) * WI + tn * 16 + col] = (_Float16)acc[r];
  }
  if (tid < 64) {   // landmark similarity (row dot gk)
    float a = 0.f;
    for (int dd = 0; dd < DHEAD; ++dd) a += (float)Qs[tid * DHEAD + dd] * gks[dd];
    simg[tid] = a;
  }
  __syncthreads();

  // masked softmax over [landmark | 192 local keys]; probs stored back as f16
  if (tid < 64) {
    float mx = simg[tid];
    for (int j = 0; j < WI; ++j) {
      int tg = tbase + j;
      if (tg >= 0 && tg < WI) mx = fmaxf(mx, (float)Sch[tid * WI + j]);
    }
    float se = __expf(simg[tid] - mx);
    float ssum = se;
    for (int j = 0; j < WI; ++j) {
      int tg = tbase + j;
      float e = (tg >= 0 && tg < WI) ? __expf((float)Sch[tid * WI + j] - mx) : 0.f;
      Sch[tid * WI + j] = (_Float16)e;
      ssum += e;
    }
    float inv = 1.f / ssum;
    pg[tid] = se * inv;
    for (int j = 0; j < WI; ++j)
      Sch[tid * WI + j] = (_Float16)((float)Sch[tid * WI + j] * inv);
  }
  __syncthreads();

  // stage V^T (d, key); window offsets are multiples of 8 -> whole chunks
  for (int ch = tid; ch < DHEAD * 24; ch += 256) {
    int dd = ch / 24, jc = ch - dd * 24;
    int t0 = tbase + jc * 8;
    v8h val = v8h_zero();
    if (t0 >= 0 && t0 < WI)
      val = *(const v8h*)(v16g + ((size_t)sh * DHEAD + dd) * WI + t0);
    *(v8h*)(VsT + dd * WI + jc * 8) = val;
  }
  __syncthreads();

  // out = P (64x192) @ V (192x32) + pg * gv; 4x2 tiles, one per wave
  {
    int tm = waveId >> 1, tn = waveId & 1;
    v8f acc = {0.f,0.f,0.f,0.f,0.f,0.f,0.f,0.f};
    for (int ks = 0; ks < 6; ++ks) {
      v16h a = load_a_frag(Sch + tm * 16 * WI + ks * 32, WI);
      v16h b = load_bt_frag(VsT + (size_t)(tn * 16) * WI + ks * 32, WI);
      acc = wmma32(a, b, acc);
    }
    int col = lane & 15, rb = (lane >> 4) * 8;
    int s = sh >> 3, h = sh & 7;
#pragma unroll
    for (int r = 0; r < 8; ++r) {
      int row = tm * 16 + rb + r;
      int t = win * WS + row;
      float o = acc[r] + pg[row] * gvs[tn * 16 + col];
      attn16[(size_t)(s * WI + t) * INNER + h * DHEAD + tn * 16 + col] = (_Float16)o;
    }
  }
}

// y (tok,128) + bo -> xcur (C,H,W) per axis mapping
__global__ __launch_bounds__(256) void k_scatter_attn(const float* __restrict__ yin,
                                                      const float* __restrict__ bo,
                                                      float* __restrict__ xout, int axis) {
  int i = blockIdx.x * 256 + threadIdx.x;
  if (i >= HW * DIMC) return;
  int tok = i >> 7, c = i & 127;
  int s = tok / WI, t = tok - s * WI;
  int hh = axis ? s : t;
  int ww = axis ? t : s;
  xout[(size_t)c * HW + hh * WI + ww] = yin[i] + bo[c];
}

// ChanLayerNorm: (x-mu)/(std+1e-5)*g+b over C, per pixel -> f16 (C,H,W)
__global__ __launch_bounds__(128) void k_chan_ln(const float* __restrict__ x,
                                                 const float* __restrict__ g,
                                                 const float* __restrict__ b,
                                                 _Float16* __restrict__ cn) {
  int p = blockIdx.x;
  int c = threadIdx.x;
  float v = x[(size_t)c * HW + p];
  __shared__ float red[128];
  red[c] = v; __syncthreads();
  for (int o = 64; o > 0; o >>= 1) { if (c < o) red[c] += red[c + o]; __syncthreads(); }
  float mu = red[0] * (1.f / 128.f);
  __syncthreads();
  float d = v - mu;
  red[c] = d * d; __syncthreads();
  for (int o = 64; o > 0; o >>= 1) { if (c < o) red[c] += red[c + o]; __syncthreads(); }
  float sd = sqrtf(red[0] * (1.f / 128.f));
  cn[(size_t)c * HW + p] = (_Float16)(d / (sd + 1e-5f) * g[c] + b[c]);
}

// hid (pix,OC) + bias -> LeakyReLU -> (OC,H,W) f16
__global__ __launch_bounds__(256) void k_lrelu_pack(const float* __restrict__ hid,
                                                    const float* __restrict__ bias,
                                                    _Float16* __restrict__ h16, int OC) {
  int i = blockIdx.x * 256 + threadIdx.x;
  if (i >= HW * OC) return;
  int p = i / OC, oc = i - p * OC;
  float v = hid[i] + bias[oc];
  v = v >= 0.f ? v : 0.01f * v;
  h16[(size_t)oc * HW + p] = (_Float16)v;
}

// yin (pix,C) + bias -> xcur (C,H,W) f32
__global__ __launch_bounds__(256) void k_bias_nchw(const float* __restrict__ yin,
                                                   const float* __restrict__ bias,
                                                   float* __restrict__ xout, int C) {
  int i = blockIdx.x * 256 + threadIdx.x;
  if (i >= HW * C) return;
  int p = i / C, c = i - p * C;
  xout[(size_t)c * HW + p] = yin[i] + bias[c];
}

// ------------------------------ host orchestration -------------------------

extern "C" void kernel_launch(void* const* d_in, const int* in_sizes, int n_in,
                              void* d_out, int out_size, void* d_ws, size_t ws_size,
                              hipStream_t stream) {
  (void)in_sizes; (void)n_in; (void)out_size; (void)ws_size;

  const float* x_in    = (const float*)d_in[0];
  const float* ln_g    = (const float*)d_in[1];
  const float* ln_b    = (const float*)d_in[2];
  const float* Wq      = (const float*)d_in[3];
  const float* Wkv     = (const float*)d_in[4];
  const float* Wproj   = (const float*)d_in[5];
  const float* lnl_g   = (const float*)d_in[6];
  const float* lnl_b   = (const float*)d_in[7];
  const float* lng_g   = (const float*)d_in[8];
  const float* lng_b   = (const float*)d_in[9];
  const float* Wo      = (const float*)d_in[10];
  const float* bo      = (const float*)d_in[11];
  const float* cln_g   = (const float*)d_in[12];
  const float* cln_b   = (const float*)d_in[13];
  const float* conv1_w = (const float*)d_in[14];
  const float* conv1_b = (const float*)d_in[15];
  const float* conv2_w = (const float*)d_in[16];
  const float* conv2_b = (const float*)d_in[17];

  char* wsb = (char*)d_ws;
  size_t off = 0;
  auto alloc = [&](size_t bytes) {
    char* p = wsb + off;
    off += (bytes + 255) & ~(size_t)255;
    return (void*)p;
  };

  float*     xcur   = (float*)alloc((size_t)HW * DIMC * 4);
  _Float16*  xn16   = (_Float16*)alloc((size_t)HW * DIMC * 2);       // also cn16
  float*     q32    = (float*)alloc((size_t)HW * INNER * 4);         // also h16 region
  float*     kv32   = (float*)alloc((size_t)HW * 512 * 4);           // also hid32
  _Float16*  q16    = (_Float16*)alloc((size_t)HW * INNER * 2);
  _Float16*  lk16   = (_Float16*)alloc((size_t)HW * INNER * 2);
  _Float16*  v16b   = (_Float16*)alloc((size_t)HW * INNER * 2);
  _Float16*  attn16 = (_Float16*)alloc((size_t)HW * INNER * 2);
  float*     y32    = (float*)alloc((size_t)HW * DIMC * 4);
  float*     gk32   = (float*)alloc((size_t)1536 * DHEAD * 4);
  float*     gv32   = (float*)alloc((size_t)1536 * DHEAD * 4);
  _Float16*  wq16   = (_Float16*)alloc((size_t)DIMC * INNER * 2);
  _Float16*  wkv16  = (_Float16*)alloc((size_t)DIMC * 512 * 2);
  _Float16*  wo16   = (_Float16*)alloc((size_t)INNER * DIMC * 2);
  _Float16*  wc16   = (_Float16*)alloc((size_t)4608 * 512 * 2);      // conv weight N x K
  _Float16*  cn16   = xn16;
  float*     hid32  = kv32;
  _Float16*  h16    = (_Float16*)q32;

  hipMemcpyAsync(xcur, x_in, (size_t)HW * DIMC * 4, hipMemcpyDeviceToDevice, stream);

  for (int l = 0; l < 2; ++l) {
    for (int a = 0; a < 2; ++a) {
      int la = l * 2 + a;
      k_gather_ln<<<HW, 128, 0, stream>>>(xcur, ln_g + la * DIMC, ln_b + la * DIMC,
                                          xn16, a);
      k_cvt_w_T<<<(DIMC * INNER + 255) / 256, 256, 0, stream>>>(
          Wq + (size_t)la * DIMC * INNER, wq16, DIMC, INNER);
      k_cvt_w_T<<<(DIMC * 512 + 255) / 256, 256, 0, stream>>>(
          Wkv + (size_t)la * DIMC * 512, wkv16, DIMC, 512);
      k_cvt_w_T<<<(INNER * DIMC + 255) / 256, 256, 0, stream>>>(
          Wo + (size_t)la * INNER * DIMC, wo16, INNER, DIMC);

      k_gemm<<<dim3(INNER / 64, HW / 64), 256, 0, stream>>>(xn16, wq16, q32,
                                                            HW, INNER, DIMC);
      k_gemm<<<dim3(512 / 64, HW / 64), 256, 0, stream>>>(xn16, wkv16, kv32,
                                                          HW, 512, DIMC);
      k_scale_q<<<(HW * INNER + 255) / 256, 256, 0, stream>>>(q32, q16);
      k_ln_k<<<HW, 256, 0, stream>>>(kv32, lnl_g + la * DHEAD, lnl_b + la * DHEAD,
                                     lk16, v16b);
      k_landmark<<<1536, 192, 0, stream>>>(kv32, Wproj + la * DHEAD,
                                           lng_g + la * DHEAD, lng_b + la * DHEAD,
                                           gk32, gv32);
      k_attn<<<dim3(NWIN, 1536), 256, 0, stream>>>(q16, lk16, v16b, gk32, gv32,
                                                   attn16);
      k_gemm<<<dim3(DIMC / 64, HW / 64), 256, 0, stream>>>(attn16, wo16, y32,
                                                           HW, DIMC, INNER);
      k_scatter_attn<<<(HW * DIMC + 255) / 256, 256, 0, stream>>>(
          y32, bo + la * DIMC, xcur, a);
    }
    // conv FFN
    k_chan_ln<<<HW, 128, 0, stream>>>(xcur, cln_g + l * DIMC, cln_b + l * DIMC, cn16);
    k_prep_conv_w<<<(1152 * 512 + 255) / 256, 256, 0, stream>>>(
        conv1_w + (size_t)l * 512 * DIMC * 9, wc16, DIMC, 512);
    k_conv_gemm<<<dim3(512 / 64, HW / 64), 256, 0, stream>>>(cn16, wc16, hid32,
                                                             DIMC, 512);
    k_lrelu_pack<<<(HW * 512 + 255) / 256, 256, 0, stream>>>(
        hid32, conv1_b + l * 512, h16, 512);
    k_prep_conv_w<<<(4608 * 128 + 255) / 256, 256, 0, stream>>>(
        conv2_w + (size_t)l * DIMC * 512 * 9, wc16, 512, DIMC);
    k_conv_gemm<<<dim3(DIMC / 64, HW / 64), 256, 0, stream>>>(h16, wc16, y32,
                                                              512, DIMC);
    k_bias_nchw<<<(HW * DIMC + 255) / 256, 256, 0, stream>>>(
        y32, conv2_b + l * DIMC, xcur, DIMC);
  }

  hipMemcpyAsync(d_out, xcur, (size_t)HW * DIMC * 4, hipMemcpyDeviceToDevice, stream);
}
